// MetricSelfAttention_82824149336108
// MI455X (gfx1250) — matrix operand order repaired and checked by reading, not compiled, not run on hardware
//
#include <hip/hip_runtime.h>
#include <hip/hip_bf16.h>

// MetricSelfAttention on gfx1250: six bf16-WMMA GEMMs.
// B=16, W=512, C=4096, N=8, k=512. Workspace use: ~196 MiB.
// Block tile 256x128, 8 waves, each wave 64x64 (4x4 WMMA 16x16x32 bf16),
// double-buffered LDS with register staging (1 barrier / K-step).

typedef __attribute__((ext_vector_type(8)))  float  v8f;
typedef __bf16 bf16_t;
typedef __attribute__((ext_vector_type(16))) bf16_t v16bf;

union BFrag { v16bf v; unsigned int u[8]; };

__device__ __forceinline__ unsigned short f32_to_bf16(float f) {
  union { float f; unsigned int u; } c; c.f = f;
  unsigned int r = c.u + 0x7FFFu + ((c.u >> 16) & 1u);   // round-nearest-even
  return (unsigned short)(r >> 16);
}

// 16 contiguous elements: global/local -> local array (vectorized).
template<typename T>
__device__ __forceinline__ void ldg16(const T* __restrict__ src, T* dst) {
  if constexpr (__is_same(T, float)) {
    const float4* s = (const float4*)src; float4* d = (float4*)dst;
#pragma unroll
    for (int q = 0; q < 4; ++q) d[q] = s[q];
  } else {
    const uint4* s = (const uint4*)src; uint4* d = (uint4*)dst;
#pragma unroll
    for (int q = 0; q < 2; ++q) d[q] = s[q];
  }
}

// 16 elements (fp32 or bf16 bits) -> 16 bf16, optional tril mask (zero col>row).
template<typename T, bool TRIL>
__device__ __forceinline__ void cvt16(const T* __restrict__ src,
                                      unsigned short* out, int row, int colbase) {
  if constexpr (__is_same(T, float)) {
#pragma unroll
    for (int e = 0; e < 16; ++e) {
      float x = src[e];
      if (TRIL && (colbase + e) > row) x = 0.0f;
      out[e] = f32_to_bf16(x);
    }
  } else {
#pragma unroll
    for (int e = 0; e < 16; ++e) {
      unsigned short u = src[e];
      if (TRIL && (colbase + e) > row) u = 0;
      out[e] = u;
    }
  }
}

constexpr int BM = 256, BN = 128, BK = 32, LDP = BK + 2; // 68B LDS row: b32-aligned pairs

// ---- staging helpers -------------------------------------------------------
template<typename AT, bool TRANS_A>
__device__ __forceinline__ void stageA_load(const AT* __restrict__ Ab, int m0, int k0,
                                            int lda, int tid, AT* st, bool pref) {
  if constexpr (!TRANS_A) {                 // A[i,k] row-major: thread owns row tid
    const AT* src = Ab + (long)(m0 + tid) * lda + k0;
    if (pref) __builtin_prefetch((const void*)(src + BK), 0, 1);
    ldg16(src, st); ldg16(src + 16, st + 16);
  } else {                                  // A[i,k] = G[k*lda+i]: read along i
    const int kr = tid & 31, cs = (tid >> 5) * 32;
    const AT* src = Ab + (long)(k0 + kr) * lda + (m0 + cs);
    if (pref) __builtin_prefetch((const void*)(src + (long)BK * lda), 0, 1);
    ldg16(src, st); ldg16(src + 16, st + 16);
  }
}

template<typename AT, bool TRANS_A, bool TRIL_A>
__device__ __forceinline__ void stageA_store(const AT* st, unsigned short (*As)[LDP],
                                             int m0, int k0, int tid) {
  unsigned short vals[32];
  if constexpr (!TRANS_A) {
    cvt16<AT, TRIL_A>(st,      vals,      m0 + tid, k0);
    cvt16<AT, TRIL_A>(st + 16, vals + 16, m0 + tid, k0 + 16);
#pragma unroll
    for (int e = 0; e < 32; ++e) As[tid][e] = vals[e];
  } else {
    const int kr = tid & 31, cs = (tid >> 5) * 32;
    cvt16<AT, false>(st,      vals,      0, 0);
    cvt16<AT, false>(st + 16, vals + 16, 0, 0);
#pragma unroll
    for (int e = 0; e < 32; ++e) As[cs + e][kr] = vals[e];
  }
}

template<typename BT, bool TRANS_B>
__device__ __forceinline__ void stageB_load(const BT* __restrict__ Bb, int n0, int k0,
                                            int ldb, int tid, BT* st, bool pref) {
  if constexpr (TRANS_B) {                  // B[k,j] = G[j*ldb+k]: contiguous along k
    const int r = tid >> 1, kc = (tid & 1) * 16;
    const BT* src = Bb + (long)(n0 + r) * ldb + (k0 + kc);
    if (pref) __builtin_prefetch((const void*)(src + BK), 0, 1);
    ldg16(src, st);
  } else {                                  // B row-major K x N: read along j
    const int kr = tid & 31, cs = (tid >> 5) * 16;
    const BT* src = Bb + (long)(k0 + kr) * ldb + (n0 + cs);
    if (pref) __builtin_prefetch((const void*)(src + (long)BK * ldb), 0, 1);
    ldg16(src, st);
  }
}

template<typename BT, bool TRANS_B, bool TRIL_BT>
__device__ __forceinline__ void stageB_store(const BT* st, unsigned short (*Bs)[LDP],
                                             int n0, int k0, int tid) {
  unsigned short vals[16];
  if constexpr (TRANS_B) {
    const int r = tid >> 1, kc = (tid & 1) * 16;
    cvt16<BT, TRIL_BT>(st, vals, n0 + r, k0 + kc);   // mask: k <= j
#pragma unroll
    for (int e = 0; e < 16; ++e) Bs[r][kc + e] = vals[e];
  } else {
    const int kr = tid & 31, cs = (tid >> 5) * 16;
    cvt16<BT, false>(st, vals, 0, 0);
#pragma unroll
    for (int e = 0; e < 16; ++e) Bs[cs + e][kr] = vals[e];
  }
}

// Generic tiled WMMA GEMM: C[i,j] = scale * sum_k A[i,k]*B[k,j] (+bias[j]) (*tril)
//  TRANS_A: A[i,k]=Ag[k*lda+i]   TRANS_B: B[k,j]=Bg[j*ldb+k]
//  TRIL_A : mask A source (k<=i) TRIL_BT: mask transposed-B source (k<=j)
//  TRIL_C : zero output j>i      HEAD_C : store to (b,w,c) head layout
//  SCATTER_P: store to (b,n,w,k) layout   A_MOD8: A batch index = z&7
template<typename AT, typename BT, typename CT,
         bool TRANS_A, bool TRANS_B, bool TRIL_A, bool TRIL_BT,
         bool TRIL_C, bool HEAD_C, bool SCATTER_P, bool A_MOD8, bool HAS_BIAS>
__global__ void __launch_bounds__(256)
gemm_wmma(const AT* __restrict__ A, const BT* __restrict__ B,
          CT* __restrict__ C, const float* __restrict__ bias,
          int Ksz, int lda, int ldb, int ldc,
          long strideA, long strideB, long strideC, float scale)
{
  __shared__ unsigned short As[2][BM][LDP];
  __shared__ unsigned short Bs[2][BN][LDP];   // [n][k]: frags read k-pairs

  const int tid  = threadIdx.x;
  const int lane = tid & 31;
  const int wv   = tid >> 5;
  const int wr   = wv & 3;      // 4 wave rows x 64
  const int wc   = wv >> 2;     // 2 wave cols x 64
  const int z    = blockIdx.z;
  const int m0   = blockIdx.y * BM;
  const int n0   = blockIdx.x * BN;

  const AT* Ab = A + (A_MOD8 ? (long)(z & 7) * strideA : (long)z * strideA);
  const BT* Bb = B + (long)z * strideB;

  v8f acc[4][4] = {};

  alignas(16) AT a_st[32];
  alignas(16) BT b_st[16];

  const int nk = Ksz / BK;

  // prologue: stage tile 0
  stageA_load<AT, TRANS_A>(Ab, m0, 0, lda, tid, a_st, nk > 1);
  stageB_load<BT, TRANS_B>(Bb, n0, 0, ldb, tid, b_st, nk > 1);
  stageA_store<AT, TRANS_A, TRIL_A>(a_st, As[0], m0, 0, tid);
  stageB_store<BT, TRANS_B, TRIL_BT>(b_st, Bs[0], n0, 0, tid);
  __syncthreads();

  for (int it = 0; it < nk; ++it) {
    const int cur = it & 1, nxt = cur ^ 1;
    const int k1 = (it + 1) * BK;

    if (it + 1 < nk) {                       // issue next tile's global loads early
      stageA_load<AT, TRANS_A>(Ab, m0, k1, lda, tid, a_st, it + 2 < nk);
      stageB_load<BT, TRANS_B>(Bb, n0, k1, ldb, tid, b_st, it + 2 < nk);
    }

    // ---- 16 WMMAs on current tile (16-bit frags, contiguous k per lane half) ----
    BFrag bfr[4];
#pragma unroll
    for (int sc = 0; sc < 4; ++sc) {
      const int n  = wc * 64 + sc * 16 + (lane & 15);
      const int kb = (lane >> 4) * 16;       // lanes 16-31: K=16..31
#pragma unroll
      for (int vv = 0; vv < 8; ++vv)
        bfr[sc].u[vv] = *(const unsigned int*)&Bs[cur][n][kb + 2 * vv];
    }
#pragma unroll
    for (int sr = 0; sr < 4; ++sr) {
      BFrag afr;
      const int m  = wr * 64 + sr * 16 + (lane & 15);
      const int kb = (lane >> 4) * 16;       // lanes 16-31: K=16..31
#pragma unroll
      for (int vv = 0; vv < 8; ++vv) {
        const int kk = kb + 2 * vv;
        afr.u[vv] = *(const unsigned int*)&As[cur][m][kk];
      }
#pragma unroll
      for (int sc = 0; sc < 4; ++sc)
        acc[sr][sc] = __builtin_amdgcn_wmma_f32_16x16x32_bf16(
            false, afr.v, false, bfr[sc].v, (short)0, acc[sr][sc], false, false);
    }

    if (it + 1 < nk) {                       // convert + store next tile, swap
      stageA_store<AT, TRANS_A, TRIL_A>(a_st, As[nxt], m0, k1, tid);
      stageB_store<BT, TRANS_B, TRIL_BT>(b_st, Bs[nxt], n0, k1, tid);
      __syncthreads();
    }
  }

  // ---- epilogue ----
#pragma unroll
  for (int sr = 0; sr < 4; ++sr) {
#pragma unroll
    for (int sc = 0; sc < 4; ++sc) {
      const long j = n0 + wc * 64 + sc * 16 + (lane & 15);
#pragma unroll
      for (int vv = 0; vv < 8; ++vv) {
        const long i = m0 + wr * 64 + sr * 16 + (lane >> 4) * 8 + vv;
        float val = acc[sr][sc][vv] * scale;
        if constexpr (HAS_BIAS) val += bias[j];
        if constexpr (TRIL_C) { if (j > i) val = 0.0f; }
        long idx;
        if constexpr (SCATTER_P) {        // (i=b*512+w, j=n*512+kk) -> (b,n,w,kk)
          const long bb = i >> 9, ww = i & 511, nn = j >> 9, kk = j & 511;
          idx = ((bb * 8 + nn) * 512 + ww) * 512 + kk;
        } else if constexpr (HEAD_C) {    // z=b*8+n, (i=w, j=kk) -> (b,w,n*512+kk)
          const long bb = z >> 3, nn = z & 7;
          idx = (bb * 512 + i) * 4096 + nn * 512 + j;
        } else {
          idx = (long)z * strideC + i * (long)ldc + j;
        }
        if constexpr (__is_same(CT, float)) C[idx] = val;
        else                                C[idx] = f32_to_bf16(val);
      }
    }
  }
}

extern "C" void kernel_launch(void* const* d_in, const int* in_sizes, int n_in,
                              void* d_out, int out_size, void* d_ws, size_t ws_size,
                              hipStream_t stream) {
  const float* x  = (const float*)d_in[0];   // (16,512,4096)
  const float* W1 = (const float*)d_in[1];   // (4096,4096)
  const float* b1 = (const float*)d_in[2];   // (4096)
  const float* pm = (const float*)d_in[3];   // (8,512,512)
  const float* W2 = (const float*)d_in[4];   // (4096,4096)
  const float* b2 = (const float*)d_in[5];   // (4096)
  float* out = (float*)d_out;                // (16,512,4096) fp32

  // workspace (bf16): P | M | MP | S ; H reuses MP (kernels run in stream order)
  unsigned short* Pbf  = (unsigned short*)d_ws;       // 16*8*512*512
  unsigned short* Mbf  = Pbf  + 33554432ul;           // 8*512*512
  unsigned short* MPbf = Mbf  + 2097152ul;            // 16*8*512*512
  unsigned short* Sbf  = MPbf + 33554432ul;           // 16*8*512*512
  unsigned short* Hbf  = MPbf;                        // reuse

  dim3 blk(256);

  // 1) M[n] = tril(pm[n]) * tril(pm[n])^T                 (8 x 512x512x512)
  gemm_wmma<float, float, unsigned short,
            false, true, true, true, false, false, false, false, false>
      <<<dim3(4, 2, 8), blk, 0, stream>>>(pm, pm, Mbf, nullptr,
          512, 512, 512, 512, 262144L, 262144L, 262144L, 1.0f);

  // 2) P = x @ W1^T + b1, scattered to (b,n,w,k)          (8192x4096x4096)
  gemm_wmma<float, float, unsigned short,
            false, true, false, false, false, false, true, false, true>
      <<<dim3(32, 32, 1), blk, 0, stream>>>(x, W1, Pbf, b1,
          4096, 4096, 4096, 0, 0L, 0L, 0L, 1.0f);

  // 3) MP[b,n] = M[n] @ P[b,n]                            (128 x 512x512x512)
  gemm_wmma<unsigned short, unsigned short, unsigned short,
            false, false, false, false, false, false, false, true, false>
      <<<dim3(4, 2, 128), blk, 0, stream>>>(Mbf, Pbf, MPbf, nullptr,
          512, 512, 512, 512, 262144L, 262144L, 262144L, 1.0f);

  // 4) S[b,n] = (P[b,n]^T @ MP[b,n]) * tril / sqrt(512)   (128 x 512x512x512)
  gemm_wmma<unsigned short, unsigned short, unsigned short,
            true, false, false, false, true, false, false, false, false>
      <<<dim3(4, 2, 128), blk, 0, stream>>>(Pbf, MPbf, Sbf, nullptr,
          512, 512, 512, 512, 262144L, 262144L, 262144L, 0.04419417382415922f);

  // 5) H = S[b,n] @ P[b,n], scattered to (b,w,c)          (128 x 512x512x512)
  gemm_wmma<unsigned short, unsigned short, unsigned short,
            false, false, false, false, false, true, false, false, false>
      <<<dim3(4, 2, 128), blk, 0, stream>>>(Sbf, Pbf, Hbf, nullptr,
          512, 512, 512, 0, 262144L, 262144L, 0L, 1.0f);

  // 6) out = H @ W2^T + b2                                (8192x4096x4096)
  gemm_wmma<unsigned short, float, float,
            false, true, false, false, false, false, false, false, true>
      <<<dim3(32, 32, 1), blk, 0, stream>>>(Hbf, W2, out, b2,
          4096, 4096, 4096, 4096, 0L, 0L, 0L, 1.0f);
}
